// QuantumAttention_65481071402019
// MI455X (gfx1250) — compile-verified, matrix operand order checked by reading, not run on hardware
//
#include <hip/hip_runtime.h>
#include <hip/hip_bf16.h>

typedef __attribute__((ext_vector_type(16))) _Float16 v16h;
typedef __attribute__((ext_vector_type(8)))  _Float16 v8h;
typedef __attribute__((ext_vector_type(8)))  float    v8f;
typedef __attribute__((ext_vector_type(2)))  float    v2f;

#define N_HEADS   16
#define N_QUBITS  8
#define DIM       256        // 2^8
#define Q_LAYERS  2
#define N_TOKENS  2048       // BATCH*SEQ
#define EMBED     128

#define WAVES_PER_BLOCK 4
// dynamic LDS layout (bytes)
#define W_LDS_BYTES     (DIM * DIM * 2)          // 131072: fp16 U_h, shared by block
#define AF_OFF          0                        // fp16 [16][256]  (8192 B)
#define BF_OFF          8192                     // fp16 [16][256]  (8192 B)
#define PR_OFF          16384                    // fp32 [16][256]  (16384 B)
#define CB_OFF          32768                    // fp32 [16][8]    (512 B)
#define SB_OFF          33280                    // fp32 [16][8]    (512 B)
#define WAVE_LDS_BYTES  33792
#define TOTAL_LDS_BYTES (W_LDS_BYTES + WAVES_PER_BLOCK * WAVE_LDS_BYTES)  // 266240

// ---------------------------------------------------------------------------
// Kernel 1: build per-head real unitary U_h (RY layers + CNOT ladders applied
// to identity), store as fp16 W[h][s_out][s_in] for use as WMMA B operand.
// ---------------------------------------------------------------------------
__global__ __launch_bounds__(256)
void qa_build_unitaries(const float* __restrict__ params,
                        float* __restrict__ Mbuf,
                        _Float16* __restrict__ W) {
    const int h = blockIdx.x;
    const int t = threadIdx.x;                  // 0..255
    float* M = Mbuf + (size_t)h * DIM * DIM;

    // identity
    for (int i = t; i < DIM * DIM; i += 256)
        M[i] = ((i >> 8) == (i & 255)) ? 1.0f : 0.0f;
    __syncthreads();

    for (int layer = 0; layer < Q_LAYERS; ++layer) {
        // --- RY(theta) on each wire (wire 0 = MSB): left-multiply row op ---
        for (int w = 0; w < N_QUBITS; ++w) {
            const float th = 0.5f * params[(h * Q_LAYERS + layer) * N_QUBITS + w];
            const float cr = cosf(th), sr = sinf(th);
            const int pos = 7 - w;
            const int pair = t & 127;           // 128 row-pairs
            const int r0 = ((pair >> pos) << (pos + 1)) | (pair & ((1 << pos) - 1));
            const int r1 = r0 | (1 << pos);
            const int cbase = (t >> 7) * 128;   // two threads split the columns
            float* row0 = M + r0 * DIM;
            float* row1 = M + r1 * DIM;
            for (int c = cbase; c < cbase + 128; ++c) {
                const float m0 = row0[c], m1 = row1[c];
                row0[c] = cr * m0 - sr * m1;
                row1[c] = sr * m0 + cr * m1;
            }
            __syncthreads();
        }
        // --- CNOT(control=w, target=w+1): swap rows |c=1,t=0> <-> |c=1,t=1> ---
        for (int w = 0; w < N_QUBITS - 1; ++w) {
            const int pt = 6 - w;               // target bit position
            const int q = t & 63;               // 64 row-pairs
            const int r0 = ((q >> pt) << (pt + 2)) | (1 << (pt + 1)) | (q & ((1 << pt) - 1));
            const int r1 = r0 | (1 << pt);
            const int cbase = (t >> 6) * 64;    // four threads split the columns
            float* row0 = M + r0 * DIM;
            float* row1 = M + r1 * DIM;
            for (int c = cbase; c < cbase + 64; ++c) {
                const float tmp = row0[c];
                row0[c] = row1[c];
                row1[c] = tmp;
            }
            __syncthreads();
        }
    }

    // convert to fp16 for WMMA B operand: W[h][s_out][s_in]
    _Float16* Wh = W + (size_t)h * DIM * DIM;
    for (int i = t; i < DIM * DIM; i += 256)
        Wh[i] = (_Float16)M[i];
}

// ---------------------------------------------------------------------------
// Kernel 2: 4 waves per block share one LDS copy of U_h; each wave owns one
// 16-token tile.
//   psi_enc = a + i b            (real vectors, product-state encoding)
//   probs   = (U a)^2 + (U b)^2  via v_wmma_f32_16x16x32_f16 (B from LDS)
//   z       = probs x signs      via v_wmma_f32_16x16x4_f32  (f32 accuracy)
// ---------------------------------------------------------------------------
__global__ __launch_bounds__(32 * WAVES_PER_BLOCK)
void qa_measure(const float* __restrict__ x,
                const _Float16* __restrict__ W,
                float* __restrict__ out) {
    extern __shared__ char smem[];
    const int h    = blockIdx.y;                 // 0..15   (head)
    const int wave = threadIdx.x >> 5;           // 0..3
    const int lane = threadIdx.x & 31;           // 0..31
    const int tile = blockIdx.x * WAVES_PER_BLOCK + wave;   // 0..127
    const int t0   = tile * 16;

    _Float16* Wlds = (_Float16*)smem;
    char* wbase = smem + W_LDS_BYTES + wave * WAVE_LDS_BYTES;
    _Float16* af    = (_Float16*)(wbase + AF_OFF);   // [16][256] Re(psi_enc)
    _Float16* bf    = (_Float16*)(wbase + BF_OFF);   // [16][256] Im(psi_enc)
    float*    probs = (float*)(wbase + PR_OFF);      // [16][256]
    float*    cbuf  = (float*)(wbase + CB_OFF);      // [16][8]
    float*    sbuf  = (float*)(wbase + SB_OFF);      // [16][8]

    // --- cooperative stage of U_h into LDS (128 KB, shared by 4 waves) ---
    {
        const v8h* src = (const v8h*)(W + (size_t)h * DIM * DIM);
        v8h* dst = (v8h*)Wlds;
        for (int i = threadIdx.x; i < DIM * DIM / 8; i += 32 * WAVES_PER_BLOCK)
            dst[i] = src[i];
    }

    // --- encoding angles: c = cos(x/2), s = sin(x/2) per (token, wire) ---
    for (int i = lane; i < 16 * N_QUBITS; i += 32) {
        const int tt = i >> 3, w = i & 7;
        const float half = 0.5f * x[(size_t)(t0 + tt) * EMBED + h * N_QUBITS + w];
        cbuf[tt * N_QUBITS + w] = cosf(half);
        sbuf[tt * N_QUBITS + w] = sinf(half);
    }
    __syncthreads();

    // --- product-state amplitudes: (-i)^popcount(s) * prod(c/s) ---
    for (int i = lane; i < 16 * DIM; i += 32) {
        const int tt = i >> 8, s = i & 255;
        float p = 1.0f;
#pragma unroll
        for (int w = 0; w < N_QUBITS; ++w)
            p *= ((s >> (7 - w)) & 1) ? sbuf[tt * N_QUBITS + w] : cbuf[tt * N_QUBITS + w];
        const int pc = __popc(s) & 3;
        const float a = (pc == 0) ? p : ((pc == 2) ? -p : 0.0f);
        const float b = (pc == 1) ? -p : ((pc == 3) ? p : 0.0f);
        af[tt * DIM + s] = (_Float16)a;
        bf[tt * DIM + s] = (_Float16)b;
    }
    __syncthreads();

    // --- GEMM1: out[t][s'] = sum_s enc[t][s] * U_h[s'][s] ---
    const int m     = lane & 15;            // row (A) / col (B) index
    const int hi    = lane >> 4;
    const int khalf = hi * 8;               // K split across lane halves

    for (int ct = 0; ct < 16; ++ct) {       // 16 output-column tiles
        v8f accA = {};
        v8f accB = {};
#pragma unroll
        for (int kt = 0; kt < 8; ++kt) {    // K = 256 in steps of 32
            const int kb = kt * 32 + khalf;
            const v8h aLo = *(const v8h*)&af[m * DIM + kb];
            const v8h aHi = *(const v8h*)&af[m * DIM + kb + 16];
            const v8h bLo = *(const v8h*)&bf[m * DIM + kb];
            const v8h bHi = *(const v8h*)&bf[m * DIM + kb + 16];
            // B fragment from LDS: B[k][n] = U_h[ct*16+n][k]
            const _Float16* bp = Wlds + (size_t)(ct * 16 + m) * DIM + kb;
            const v8h wLo = *(const v8h*)bp;
            const v8h wHi = *(const v8h*)(bp + 16);
            v16h Aa, Ab, Bf;
#pragma unroll
            for (int i = 0; i < 8; ++i) {
                Aa[i] = aLo[i]; Aa[i + 8] = aHi[i];
                Ab[i] = bLo[i]; Ab[i + 8] = bHi[i];
                Bf[i] = wLo[i]; Bf[i + 8] = wHi[i];
            }
            accA = __builtin_amdgcn_wmma_f32_16x16x32_f16(
                false, Aa, false, Bf, (short)0, accA, false, false);
            accB = __builtin_amdgcn_wmma_f32_16x16x32_f16(
                false, Ab, false, Bf, (short)0, accB, false, false);
        }
        // probs tile: C/D layout -> VGPR r holds M=r (+8 for upper lanes), N=lane&15
#pragma unroll
        for (int r = 0; r < 8; ++r) {
            const float re = accA[r], im = accB[r];
            probs[(r + 8 * hi) * DIM + ct * 16 + m] = re * re + im * im;
        }
    }
    __syncthreads();

    // --- GEMM2: z[t][w] = sum_s' probs[t][s'] * sign_w(s')  (f32 WMMA) ---
    const int n = lane & 15;                // output column (wire w for n<8)
    const int shift = 7 - (n & 7);
    v8f zacc = {};
#pragma unroll 8
    for (int k2 = 0; k2 < 64; ++k2) {       // K = 256 in steps of 4
        const int kb2 = k2 * 4 + 2 * hi;    // this lane's K pair
        const v2f pA = *(const v2f*)&probs[m * DIM + kb2];
        v2f pB;
#pragma unroll
        for (int j = 0; j < 2; ++j) {
            const int sp = kb2 + j;
            const float sg = ((sp >> shift) & 1) ? -1.0f : 1.0f;
            pB[j] = (n < N_QUBITS) ? sg : 0.0f;
        }
        zacc = __builtin_amdgcn_wmma_f32_16x16x4_f32(
            false, pA, false, pB, (short)0, zacc, false, false);
    }

    if (n < N_QUBITS) {
#pragma unroll
        for (int r = 0; r < 8; ++r) {
            const int tt = r + 8 * hi;
            out[(size_t)(t0 + tt) * EMBED + h * N_QUBITS + n] = zacc[r];
        }
    }
}

// ---------------------------------------------------------------------------
extern "C" void kernel_launch(void* const* d_in, const int* in_sizes, int n_in,
                              void* d_out, int out_size, void* d_ws, size_t ws_size,
                              hipStream_t stream) {
    const float* x      = (const float*)d_in[0];   // [2,1024,128] fp32
    const float* params = (const float*)d_in[1];   // [16,2,8]     fp32
    float* out          = (float*)d_out;           // [2,1024,128] fp32

    // scratch layout: fp32 M build area (4 MB) then fp16 W (2 MB)
    float*    Mbuf = (float*)d_ws;
    _Float16* W    = (_Float16*)((char*)d_ws + (size_t)N_HEADS * DIM * DIM * sizeof(float));

    qa_build_unitaries<<<dim3(N_HEADS), dim3(256), 0, stream>>>(params, Mbuf, W);

    dim3 grid(N_TOKENS / 16 / WAVES_PER_BLOCK, N_HEADS);   // 32 x 16 blocks
    qa_measure<<<grid, dim3(32 * WAVES_PER_BLOCK), TOTAL_LDS_BYTES, stream>>>(x, W, out);
}
